// AdaDROLoss_20736102105568
// MI455X (gfx1250) — compile-verified, matrix operand order checked by reading, not run on hardware
//
#include <hip/hip_runtime.h>
#include <hip/hip_bf16.h>

// ---------------- problem constants ----------------
#define NN 2048   // ptr samples
#define MM 4096   // nu samples
#define DD 512    // feature dim
#define KK 1000   // classes
#define K2 1024   // packed GEMM K (features ++ label-embedding)

constexpr float kLambda   = 0.1f;
constexpr float kEpsilon  = 0.5f;
constexpr float kEpsSmall = 1e-8f;
constexpr float kInvDen   = 1.0f / (kLambda * kEpsilon + kEpsSmall);

// ---------------- workspace layout (bytes, all 256-aligned) ----------------
constexpr size_t OFF_WN   = 0;                                  // f32 [KK, DD]
constexpr size_t SZ_WN    = (size_t)KK * DD * 4;                // 2,048,000
constexpr size_t OFF_AHI  = OFF_WN + SZ_WN;                     // bf16 [NN, K2]
constexpr size_t SZ_A     = (size_t)NN * K2 * 2;                // 4,194,304
constexpr size_t OFF_ALO  = OFF_AHI + SZ_A;
constexpr size_t OFF_BHI  = OFF_ALO + SZ_A;                     // bf16 [MM, K2]
constexpr size_t SZ_B     = (size_t)MM * K2 * 2;                // 8,388,608
constexpr size_t OFF_BLO  = OFF_BHI + SZ_B;
constexpr size_t OFF_NUL  = OFF_BLO + SZ_B;                     // f32 [MM]
constexpr size_t OFF_E    = OFF_NUL + 16384;                    // f32 [NN, MM]
constexpr size_t OFF_RMAX = OFF_E + (size_t)NN * MM * 4;        // f32 [NN]
constexpr size_t OFF_RINV = OFF_RMAX + 8192;                    // f32 [NN]  (1/(rowsum+eps))
constexpr size_t OFF_PP   = OFF_RINV + 8192;                    // f32 [MM]
constexpr size_t OFF_CSA  = OFF_PP + 16384;                     // f32 [K2]
constexpr size_t OFF_CSB  = OFF_CSA + 4096;                     // f32 [K2]
// total ~60.8 MB

// ---------------- vector types for WMMA ----------------
typedef __attribute__((ext_vector_type(16))) __bf16        bf16x16;
typedef __attribute__((ext_vector_type(8)))  float         f32x8;
typedef __attribute__((ext_vector_type(4)))  unsigned int  u32x4;

union Frag { bf16x16 v; u32x4 q[2]; };

// CDNA5 async memory->LDS copy (16B per lane), tracked by ASYNCcnt.
// VDST VGPR = workgroup-relative LDS byte address; INST_OFFSET applies to both sides.
__device__ __forceinline__ void async_copy_b128(unsigned lds_off, const void* gptr) {
    asm volatile("global_load_async_to_lds_b128 %0, %1, off"
                 :: "v"(lds_off), "v"((unsigned long long)gptr) : "memory");
}
__device__ __forceinline__ void async_copy_b128_o16(unsigned lds_off, const void* gptr) {
    asm volatile("global_load_async_to_lds_b128 %0, %1, off offset:16"
                 :: "v"(lds_off), "v"((unsigned long long)gptr) : "memory");
}
__device__ __forceinline__ unsigned lds_addr_of(const void* p) {
    return (unsigned)(uintptr_t)p;   // generic LDS pointer: low 32 bits = LDS offset
}

// ---------------- block reductions (blockDim.x == 256) ----------------
__device__ __forceinline__ float block_reduce_sum(float v, float* sbuf) {
    int t = threadIdx.x;
    __syncthreads();
    sbuf[t] = v; __syncthreads();
    for (int s = 128; s > 0; s >>= 1) { if (t < s) sbuf[t] += sbuf[t + s]; __syncthreads(); }
    return sbuf[0];
}
__device__ __forceinline__ float block_reduce_max(float v, float* sbuf) {
    int t = threadIdx.x;
    __syncthreads();
    sbuf[t] = v; __syncthreads();
    for (int s = 128; s > 0; s >>= 1) { if (t < s) sbuf[t] = fmaxf(sbuf[t], sbuf[t + s]); __syncthreads(); }
    return sbuf[0];
}

// ---------------- kernel 1: normalize classifier rows ----------------
__global__ void k_norm_w(const float* __restrict__ W, float* __restrict__ Wn) {
    __shared__ float sbuf[256];
    int k = blockIdx.x, t = threadIdx.x;
    const float* row = W + (size_t)k * DD;
    float ss = 0.f;
    for (int c = t; c < DD; c += 256) { float v = row[c]; ss += v * v; }
    ss = block_reduce_sum(ss, sbuf);
    float inv = 1.0f / fmaxf(sqrtf(ss), kEpsSmall);
    for (int c = t; c < DD; c += 256) Wn[(size_t)k * DD + c] = row[c] * inv;
}

// ---------------- kernel 2: build packed split-bf16 operand ----------------
// row n: [ norm(feats[n]) (DD)  ||  Wn[labels[n]] (DD) ]  -> hi/lo bf16 planes
__global__ void k_build(const float* __restrict__ feats, const int* __restrict__ labels,
                        const float* __restrict__ Wn,
                        __hip_bfloat16* __restrict__ hiP, __hip_bfloat16* __restrict__ loP) {
    __shared__ float sbuf[256];
    int n = blockIdx.x, t = threadIdx.x;
    const float* row = feats + (size_t)n * DD;
    float ss = 0.f;
    for (int c = t; c < DD; c += 256) { float v = row[c]; ss += v * v; }
    ss = block_reduce_sum(ss, sbuf);
    float inv = 1.0f / fmaxf(sqrtf(ss), kEpsSmall);
    int lbl = labels[n];
    const float* emb = Wn + (size_t)lbl * DD;
    size_t base = (size_t)n * K2;
    for (int c = t; c < DD; c += 256) {
        float a = row[c] * inv;
        __hip_bfloat16 h = __float2bfloat16(a);
        hiP[base + c] = h;
        loP[base + c] = __float2bfloat16(a - __bfloat162float(h));
        float b = emb[c];
        __hip_bfloat16 h2 = __float2bfloat16(b);
        hiP[base + DD + c] = h2;
        loP[base + DD + c] = __float2bfloat16(b - __bfloat162float(h2));
    }
}

// ---------------- kernel 3: per-row CE loss on nu ----------------
__global__ void k_nu_loss(const float* __restrict__ logits, const int* __restrict__ labels,
                          float* __restrict__ losses) {
    __shared__ float sbuf[256];
    int m = blockIdx.x, t = threadIdx.x;
    const float* row = logits + (size_t)m * KK;
    float mx = -__builtin_inff();
    for (int c = t; c < KK; c += 256) mx = fmaxf(mx, row[c]);
    mx = block_reduce_max(mx, sbuf);
    float s = 0.f;
    for (int c = t; c < KK; c += 256) s += expf(row[c] - mx);
    s = block_reduce_sum(s, sbuf);
    if (t == 0) losses[m] = logf(s) + mx - row[labels[m]];
}

// ---------------- kernel 4: WMMA GEMM + fused DRO-exponent epilogue ----------------
// E[n,m] = (nu_losses[m] - lambda * (2 - A.Bt)) * invDen
// block tile: 32 (N) x 64 (M), 8 wave32's in 2x4 grid, K-chunks of 32.
// Double-buffered LDS staging via GLOBAL_LOAD_ASYNC_TO_LDS_B128 (ASYNCcnt):
// async loads complete in order, so "s_wait_asynccnt 3" waits only for the
// current chunk while the next chunk's DMA overlaps the WMMAs.
__global__ void k_gemm_exponent(const __hip_bfloat16* __restrict__ Ahi, const __hip_bfloat16* __restrict__ Alo,
                                const __hip_bfloat16* __restrict__ Bhi, const __hip_bfloat16* __restrict__ Blo,
                                const float* __restrict__ nuL, float* __restrict__ E) {
    __shared__ __attribute__((aligned(16))) unsigned short At[2][2][32][32]; // [buf][plane][row][k]
    __shared__ __attribute__((aligned(16))) unsigned short Bt[2][2][64][32];

    int t = threadIdx.x;
    int nBase = blockIdx.y * 32;
    int mBase = blockIdx.x * 64;
    int wave = t >> 5, lane = t & 31;
    int wn = wave >> 2, wm = wave & 3;     // wave tile: rows [16*wn,+16), cols [16*wm,+16)
    int m16 = lane & 15, hi = lane >> 4;

    // staging indices: A tile = 2 planes * 32 rows * 32 halfs (1 x b128 / thread)
    int ap = t >> 7;            // plane
    int ar = (t >> 2) & 31;     // row 0..31
    int ac = (t & 3) * 8;       // col (halfs)
    // B tile = 2 planes * 64 rows * 32 halfs (2 x b128 / thread)
    int bp = t >> 7;
    int bidx = t & 127;
    int br = bidx >> 1;
    int bc = (bidx & 1) * 16;

    const __hip_bfloat16* aSrcBase = (ap ? Alo : Ahi) + (size_t)(nBase + ar) * K2 + ac;
    const __hip_bfloat16* bSrcBase = (bp ? Blo : Bhi) + (size_t)(mBase + br) * K2 + bc;

    auto issue = [&](int buf, int kc) {
        async_copy_b128(lds_addr_of(&At[buf][ap][ar][ac]), aSrcBase + kc);
        unsigned ldsB = lds_addr_of(&Bt[buf][bp][br][bc]);
        async_copy_b128(ldsB, bSrcBase + kc);
        async_copy_b128_o16(ldsB, bSrcBase + kc);   // offset:16 applies to LDS and memory
    };

    f32x8 acc = {0.f, 0.f, 0.f, 0.f, 0.f, 0.f, 0.f, 0.f};

    issue(0, 0);   // prologue: chunk 0 into buffer 0 (3 async ops outstanding)

    for (int kc = 0; kc < K2; kc += 32) {
        int cur = (kc >> 5) & 1;
        bool more = (kc + 32) < K2;
        if (more) issue(cur ^ 1, kc + 32);   // overlap next chunk's DMA with compute
        if (more) { asm volatile("s_wait_asynccnt 0x3" ::: "memory"); }
        else      { asm volatile("s_wait_asynccnt 0x0" ::: "memory"); }
        __syncthreads();   // all waves' copies of `cur` are in LDS

        // fragment loads per the 16-bit 16x32 A layout (ISA 7.12.2):
        // lane L: row = L%16; VGPR 0..3 hold K = 8*hi + [0..7]; VGPR 4..7 hold K = 16 + 8*hi + [0..7]
        int arow = wn * 16 + m16;
        int brow = wm * 16 + m16;
        int o0 = 8 * hi, o1 = 16 + 8 * hi;
        Frag aHi, aLo, bHi, bLo;
        aHi.q[0] = *(const u32x4*)&At[cur][0][arow][o0];
        aHi.q[1] = *(const u32x4*)&At[cur][0][arow][o1];
        aLo.q[0] = *(const u32x4*)&At[cur][1][arow][o0];
        aLo.q[1] = *(const u32x4*)&At[cur][1][arow][o1];
        bHi.q[0] = *(const u32x4*)&Bt[cur][0][brow][o0];
        bHi.q[1] = *(const u32x4*)&Bt[cur][0][brow][o1];
        bLo.q[0] = *(const u32x4*)&Bt[cur][1][brow][o0];
        bLo.q[1] = *(const u32x4*)&Bt[cur][1][brow][o1];

        // split-bf16 product: hi*hi + hi*lo + lo*hi (f32 accumulate) ~ fp32 GEMM
        acc = __builtin_amdgcn_wmma_f32_16x16x32_bf16(false, aHi.v, false, bHi.v, (short)0, acc, false, false);
        acc = __builtin_amdgcn_wmma_f32_16x16x32_bf16(false, aHi.v, false, bLo.v, (short)0, acc, false, false);
        acc = __builtin_amdgcn_wmma_f32_16x16x32_bf16(false, aLo.v, false, bHi.v, (short)0, acc, false, false);

        __syncthreads();   // all waves done reading `cur` before it is overwritten
    }

    // epilogue: C/D layout (ISA 7.12.2): VGPR r -> row (r + 8*hi), col = lane%16
    int mg = mBase + wm * 16 + m16;
    float nl = nuL[mg];
    #pragma unroll
    for (int r = 0; r < 8; ++r) {
        int ng = nBase + wn * 16 + r + 8 * hi;
        float ctot = 2.0f - acc[r];                   // (1-cosX) + (1-cosY)
        E[(size_t)ng * MM + mg] = (nl - kLambda * ctot) * kInvDen;
    }
}

// ---------------- kernel 5: per-row max / reciprocal of sum-exp ----------------
__global__ void k_row_stats(const float* __restrict__ E, float* __restrict__ rmax, float* __restrict__ rinv) {
    __shared__ float sbuf[256];
    int n = blockIdx.x, t = threadIdx.x;
    const float* row = E + (size_t)n * MM;
    float mx = -__builtin_inff();
    for (int m = t; m < MM; m += 256) mx = fmaxf(mx, row[m]);
    mx = block_reduce_max(mx, sbuf);
    float s = 0.f;
    for (int m = t; m < MM; m += 256) s += expf(row[m] - mx);
    s = block_reduce_sum(s, sbuf);
    if (t == 0) { rmax[n] = mx; rinv[n] = 1.0f / (s + kEpsSmall); }
}

// ---------------- kernel 6: column mean of row-normalized weights ----------------
__global__ void k_col_probs(const float* __restrict__ E, const float* __restrict__ rmax,
                            const float* __restrict__ rinv, float* __restrict__ probs_pre) {
    int m = blockIdx.x * 256 + threadIdx.x;
    float acc = 0.f;
    for (int n = 0; n < NN; ++n) {
        float e = E[(size_t)n * MM + m];
        acc += expf(e - rmax[n]) * rinv[n];
    }
    probs_pre[m] = acc / (float)NN;
}

// ---------------- kernel 7: column sums of A and B (for exact mean-cos scalars) ----------------
__global__ void k_colsums(const __hip_bfloat16* __restrict__ Ahi, const __hip_bfloat16* __restrict__ Alo,
                          const __hip_bfloat16* __restrict__ Bhi, const __hip_bfloat16* __restrict__ Blo,
                          float* __restrict__ csA, float* __restrict__ csB) {
    int id = blockIdx.x * 256 + threadIdx.x;   // 0..2047
    if (id < K2) {
        int c = id; float s = 0.f;
        for (int n = 0; n < NN; ++n) {
            size_t i = (size_t)n * K2 + c;
            s += __bfloat162float(Ahi[i]) + __bfloat162float(Alo[i]);
        }
        csA[c] = s;
    } else {
        int c = id - K2; float s = 0.f;
        for (int m = 0; m < MM; ++m) {
            size_t i = (size_t)m * K2 + c;
            s += __bfloat162float(Bhi[i]) + __bfloat162float(Blo[i]);
        }
        csB[c] = s;
    }
}

// ---------------- kernel 8: finalize all outputs ----------------
__global__ void k_finish(const float* __restrict__ probs_pre, const float* __restrict__ nuL,
                         const float* __restrict__ csA, const float* __restrict__ csB,
                         float* __restrict__ out) {
    __shared__ float sbuf[256];
    __shared__ float s_tot;
    int t = threadIdx.x;
    float s = 0.f;
    for (int m = t; m < MM; m += 256) s += probs_pre[m];
    s = block_reduce_sum(s, sbuf);
    if (t == 0) s_tot = s;
    __syncthreads();
    float inv = 1.0f / (s_tot + kEpsSmall);
    float d = 0.f;
    for (int m = t; m < MM; m += 256) {
        float p = probs_pre[m] * inv;
        out[2 + m] = p;
        d += p * nuL[m];
    }
    d = block_reduce_sum(d, sbuf);
    // mean(C_X) = 1 - (sum_n a_n).(sum_m b_m)/(N*M) over feature half; same for label half
    float dx = 0.f, dy = 0.f;
    for (int c = t; c < DD; c += 256) {
        dx += csA[c] * csB[c];
        dy += csA[DD + c] * csB[DD + c];
    }
    dx = block_reduce_sum(dx, sbuf);
    dy = block_reduce_sum(dy, sbuf);
    if (t == 0) {
        float scale = 1.0f / ((float)NN * (float)MM);
        float mcx = 1.0f - dx * scale;
        float mcy = 1.0f - dy * scale;
        out[0] = d;               // total_loss (semantic part is zero)
        out[1] = d;               // dro_loss
        out[2 + MM] = mcx + mcy;  // mean(C_total)
        out[3 + MM] = mcx;        // mean(C_X)
        out[4 + MM] = mcy;        // mean(C_Y)
    }
}

// ---------------- launcher ----------------
extern "C" void kernel_launch(void* const* d_in, const int* in_sizes, int n_in,
                              void* d_out, int out_size, void* d_ws, size_t ws_size,
                              hipStream_t stream) {
    const float* ptr_features = (const float*)d_in[0];
    /* d_in[1] = ptr_logits: unused by the reference */
    const int*   ptr_labels   = (const int*)d_in[2];
    const float* nu_features  = (const float*)d_in[3];
    const float* nu_logits    = (const float*)d_in[4];
    const int*   nu_labels    = (const int*)d_in[5];
    const float* cls_w        = (const float*)d_in[6];
    float* out = (float*)d_out;

    char* ws = (char*)d_ws;
    float*          Wn   = (float*)(ws + OFF_WN);
    __hip_bfloat16* Ahi  = (__hip_bfloat16*)(ws + OFF_AHI);
    __hip_bfloat16* Alo  = (__hip_bfloat16*)(ws + OFF_ALO);
    __hip_bfloat16* Bhi  = (__hip_bfloat16*)(ws + OFF_BHI);
    __hip_bfloat16* Blo  = (__hip_bfloat16*)(ws + OFF_BLO);
    float*          nuL  = (float*)(ws + OFF_NUL);
    float*          E    = (float*)(ws + OFF_E);
    float*          rmax = (float*)(ws + OFF_RMAX);
    float*          rinv = (float*)(ws + OFF_RINV);
    float*          pp   = (float*)(ws + OFF_PP);
    float*          csA  = (float*)(ws + OFF_CSA);
    float*          csB  = (float*)(ws + OFF_CSB);

    k_norm_w      <<<KK, 256, 0, stream>>>(cls_w, Wn);
    k_build       <<<NN, 256, 0, stream>>>(ptr_features, ptr_labels, Wn, Ahi, Alo);
    k_build       <<<MM, 256, 0, stream>>>(nu_features, nu_labels, Wn, Bhi, Blo);
    k_nu_loss     <<<MM, 256, 0, stream>>>(nu_logits, nu_labels, nuL);
    k_gemm_exponent<<<dim3(MM / 64, NN / 32), 256, 0, stream>>>(Ahi, Alo, Bhi, Blo, nuL, E);
    k_row_stats   <<<NN, 256, 0, stream>>>(E, rmax, rinv);
    k_col_probs   <<<MM / 256, 256, 0, stream>>>(E, rmax, rinv, pp);
    k_colsums     <<<(2 * K2) / 256, 256, 0, stream>>>(Ahi, Alo, Bhi, Blo, csA, csB);
    k_finish      <<<1, 256, 0, stream>>>(pp, nuL, csA, csB, out);
}
